// DSDCrystal_10995116278145
// MI455X (gfx1250) — matrix operand order repaired and checked by reading, not compiled, not run on hardware
//
#include <hip/hip_runtime.h>
#include <hip/hip_bf16.h>

// ---- problem constants (match reference) -----------------------------------
#define N_NODES 25000
#define N_EDGES 250000
#define DIM     64
#define HEADS   4
#define K2      128          // 2*DIM (GEMM K)
#define NOUT    256          // HEADS*DIM (GEMM N)
#define BN_EPS  1e-5f
#define NTILES  (N_EDGES / 16)   // 15625 tiles of 16 edges (exact)
#define WAVES_PER_BLOCK 8
#define BLOCK   256
#define W_CHUNKS 2048        // 16 nt * 4 kk * 32 lanes (each chunk = 16 bf16)

typedef __bf16  bf16_t;
typedef __bf16  v16bf __attribute__((ext_vector_type(16)));
typedef float   v8f   __attribute__((ext_vector_type(8)));
typedef unsigned int u32x4 __attribute__((ext_vector_type(4)));
typedef int          i32x8 __attribute__((ext_vector_type(8)));
typedef int          i32x4 __attribute__((ext_vector_type(4)));

__device__ __forceinline__ float softplus_f(float x) {
    // numerically stable softplus
    return fmaxf(x, 0.f) + __logf(1.f + __expf(-fabsf(x)));
}

__device__ __forceinline__ v8f wmma_bf16(v16bf a, v16bf b, v8f c) {
    // D = A(16x32 bf16) * B(32x16 bf16) + C(16x16 f32)
    return __builtin_amdgcn_wmma_f32_16x16x32_bf16(
        false, a, false, b, (short)0, c, false, false);
}

// --- Tensor Data Mover: one 64 KB contiguous global->LDS DMA ----------------
// D# per CDNA5 ISA: 2-D tensor, data_size=4B,
// tile/tensor = 4096 x 4 dwords (= 64 KB), stride0 = 4096 dwords.
// This toolchain exposes the 6-arg builtin (g0, g1, g2, g3, g4, cpol).
__device__ __forceinline__ void tdm_copy_64k(const void* gsrc, void* lds_dst) {
    unsigned long long ga = (unsigned long long)(uintptr_t)gsrc;
    unsigned int ldsoff = (unsigned int)(uintptr_t)lds_dst; // flat LDS addr[31:0] = LDS offset
    u32x4 g0;
    g0[0] = 1u;                                             // count=1 (valid), user D#
    g0[1] = ldsoff;                                         // lds_addr (bytes)
    g0[2] = (unsigned int)(ga & 0xFFFFFFFFu);               // global_addr[31:0]
    g0[3] = (unsigned int)((ga >> 32) & 0x01FFFFFFu)        // global_addr[56:32]
          | (2u << 30);                                     // type = 2 ("image")
    i32x8 g1;
    g1[0] = (int)(2u << 16);        // wg_mask=0 | data_size=2 (4 bytes)
    g1[1] = (int)(4096u << 16);     // tensor_dim0[15:0] = 4096
    g1[2] = (int)(4u << 16);        // tensor_dim0[31:16]=0 | tensor_dim1[15:0]=4
    g1[3] = (int)(4096u << 16);     // tensor_dim1[31:16]=0 | tile_dim0 = 4096
    g1[4] = 4;                      // tile_dim1 = 4 | tile_dim2 = 0
    g1[5] = 4096;                   // tensor_dim0_stride[31:0] = 4096
    g1[6] = 0;                      // stride0 hi | stride1 lo (unused for 2D)
    g1[7] = 0;
    i32x4 z4 = {0, 0, 0, 0};        // groups 2/3: NULL (tensor <= 2D)
    i32x8 z8 = {0, 0, 0, 0, 0, 0, 0, 0};
    __builtin_amdgcn_tensor_load_to_lds(g0, g1, z4, z4, z8, 0);
}

// Stage pre-swizzled bf16 W from workspace into LDS with the TDM.
// Wave 0 issues the DMA and waits on TENSORcnt; barrier publishes to the WG.
__device__ __forceinline__ void stage_W_tdm(const bf16_t* __restrict__ wswz,
                                            v16bf* wlds) {
    if ((threadIdx.x >> 5) == 0) {
        tdm_copy_64k((const void*)wswz, (void*)wlds);
        __builtin_amdgcn_s_wait_tensorcnt(0);
    }
    __syncthreads();
}

// Build A fragments (16x32 bf16 per kk-step, K total = 128) for one 16-edge
// tile. Row = lane&15. Per ISA layout: lanes 0-15 hold K kb..kb+7 (elems 0-7)
// and kb+16..kb+23 (elems 8-15) with kb = kk*32; lanes 16-31 use kb+8.
// K<64 -> x[node], K>=64 -> edge_attr[e] (concat is contiguous per source).
__device__ __forceinline__ void build_afrag(const float* __restrict__ x,
                                            const float* __restrict__ edge_attr,
                                            int node, int e, int lane,
                                            v16bf a[4]) {
    int half8 = (lane & 16) ? 8 : 0;
    #pragma unroll
    for (int kk = 0; kk < 4; ++kk) {
        const float* src = (kk < 2)
            ? (x + (size_t)node * DIM + kk * 32 + half8)
            : (edge_attr + (size_t)e * DIM + (kk - 2) * 32 + half8);
        float4 c0 = ((const float4*)src)[0];   // K = kb .. kb+3
        float4 c1 = ((const float4*)src)[1];   // K = kb+4 .. kb+7
        float4 c2 = ((const float4*)src)[4];   // K = kb+16 .. kb+19
        float4 c3 = ((const float4*)src)[5];   // K = kb+20 .. kb+23
        a[kk][0]  = (bf16_t)c0.x; a[kk][1]  = (bf16_t)c0.y;
        a[kk][2]  = (bf16_t)c0.z; a[kk][3]  = (bf16_t)c0.w;
        a[kk][4]  = (bf16_t)c1.x; a[kk][5]  = (bf16_t)c1.y;
        a[kk][6]  = (bf16_t)c1.z; a[kk][7]  = (bf16_t)c1.w;
        a[kk][8]  = (bf16_t)c2.x; a[kk][9]  = (bf16_t)c2.y;
        a[kk][10] = (bf16_t)c2.z; a[kk][11] = (bf16_t)c2.w;
        a[kk][12] = (bf16_t)c3.x; a[kk][13] = (bf16_t)c3.y;
        a[kk][14] = (bf16_t)c3.z; a[kk][15] = (bf16_t)c3.w;
    }
}

// ---- kernel P: one-time swizzle of W into bf16 WMMA-B fragment layout ------
// chunk c = (nt,kk,lane): 16 contiguous bf16; col = nt*16+(lane&15),
// K base = kk*32 + (lane&16 ? 8 : 0). Lanes 0-15 hold K 0..7,16..23; 16-31 K 8..15,24..31.
__global__ __launch_bounds__(BLOCK) void k_prep(const float* __restrict__ W,
                                                bf16_t* __restrict__ wswz) {
    int c = blockIdx.x * BLOCK + threadIdx.x;
    if (c >= W_CHUNKS) return;
    int lane = c & 31;
    int kk   = (c >> 5) & 3;
    int nt   = c >> 7;
    int col  = nt * 16 + (lane & 15);
    int kb   = kk * 32 + ((lane & 16) ? 8 : 0);
    #pragma unroll
    for (int q = 0; q < 16; ++q)
        wswz[c * 16 + q] = (bf16_t)W[(kb + q + (q & 8)) * NOUT + col];
}

// ---- kernel 0: out <- bias, zero segment max/denominator -------------------
__global__ __launch_bounds__(BLOCK) void k_init(float* __restrict__ out,
                                                const float* __restrict__ bias,
                                                int* __restrict__ amaxBits,
                                                float* __restrict__ denom) {
    int i = blockIdx.x * BLOCK + threadIdx.x;
    if (i < N_NODES * DIM)  out[i] = bias[i & (DIM - 1)];
    if (i < N_NODES * HEADS) { amaxBits[i] = 0; denom[i] = 0.f; }
}

// ---- kernel 1: alpha = softplus(BN(att . [sp(cat@W)])), seg max --
__global__ __launch_bounds__(BLOCK) void k_alpha(
    const float* __restrict__ x, const float* __restrict__ edge_attr,
    const bf16_t* __restrict__ wswz, const float* __restrict__ att,
    const float* __restrict__ bn_gamma, const float* __restrict__ bn_beta,
    const float* __restrict__ bn_mean, const float* __restrict__ bn_var,
    const int* __restrict__ edge_index,
    float* __restrict__ alphaBuf, int* __restrict__ amaxBits)
{
    __shared__ v16bf wlds[W_CHUNKS];        // 64 KB bf16 W, B-frag swizzled
    stage_W_tdm(wswz, wlds);

    int wave = threadIdx.x >> 5, lane = threadIdx.x & 31;
    int tile = blockIdx.x * WAVES_PER_BLOCK + wave;
    if (tile >= NTILES) return;
    int base = tile * 16;
    int m = lane & 15;
    int e = base + m;
    int ni = edge_index[e];                // idx_i (flow target_to_source)
    int nj = edge_index[N_EDGES + e];      // idx_j

    // prefetch next tile's edge_attr rows while we compute
    if (e + 16 < N_EDGES)
        __builtin_prefetch(edge_attr + (size_t)(e + 16) * DIM, 0, 1);

    v16bf ai[4], aj[4];
    build_afrag(x, edge_attr, ni, e, lane, ai);
    build_afrag(x, edge_attr, nj, e, lane, aj);

    #pragma unroll
    for (int h = 0; h < HEADS; ++h) {      // one head = 4 N-tiles of 16 cols
        float aacc[8];
        #pragma unroll
        for (int r = 0; r < 8; ++r) aacc[r] = 0.f;
        #pragma unroll
        for (int t = 0; t < 4; ++t) {
            int nt = h * 4 + t;
            v8f ci = {}; v8f cj = {};
            #pragma unroll
            for (int kk = 0; kk < 4; ++kk) {
                v16bf b = wlds[(nt * 4 + kk) * 32 + lane];
                ci = wmma_bf16(ai[kk], b, ci);
                cj = wmma_bf16(aj[kk], b, cj);
            }
            int d  = (nt * 16 + (lane & 15)) & (DIM - 1);
            float ca = att[h * K2 + d];         // coef for out_i part
            float cb = att[h * K2 + DIM + d];   // coef for out_j part
            #pragma unroll
            for (int r = 0; r < 8; ++r)
                aacc[r] += softplus_f(ci[r]) * ca + softplus_f(cj[r]) * cb;
        }
        // reduce the 16 columns held by each 16-lane half (rows r vs r+8)
        #pragma unroll
        for (int r = 0; r < 8; ++r) {
            #pragma unroll
            for (int off = 1; off <= 8; off <<= 1)
                aacc[r] += __shfl_xor(aacc[r], off, 32);
        }
        if ((lane & 15) == 0) {
            int rowbase = (lane >> 4) * 8;     // lane0 -> rows 0..7, lane16 -> 8..15
            float scale = bn_gamma[h] * rsqrtf(bn_var[h] + BN_EPS);
            float mean = bn_mean[h], beta = bn_beta[h];
            #pragma unroll
            for (int r = 0; r < 8; ++r) {
                int em = base + rowbase + r;
                float aval = softplus_f((aacc[r] - mean) * scale + beta);
                alphaBuf[em * HEADS + h] = aval;
                int node = edge_index[em];
                // alpha >= 0 after softplus: int-max == float-max on bit patterns
                atomicMax(&amaxBits[node * HEADS + h], __float_as_int(aval));
            }
        }
    }
}

// ---- kernel 2: e = exp(alpha - amax), denom += e ---------------------------
__global__ __launch_bounds__(BLOCK) void k_norm(
    const int* __restrict__ edge_index,
    float* __restrict__ alphaBuf, const int* __restrict__ amaxBits,
    float* __restrict__ denom)
{
    int tid = blockIdx.x * BLOCK + threadIdx.x;
    if (tid >= N_EDGES * HEADS) return;
    int e = tid >> 2, h = tid & 3;
    int node = edge_index[e];
    float amax = __int_as_float(amaxBits[node * HEADS + h]);
    float ex = __expf(alphaBuf[tid] - amax);
    alphaBuf[tid] = ex;
    atomicAdd(&denom[node * HEADS + h], ex);
}

// ---- kernel 3: recompute out_j, weight, head-sum, scatter-add --------------
__global__ __launch_bounds__(BLOCK) void k_aggr(
    const float* __restrict__ x, const float* __restrict__ edge_attr,
    const bf16_t* __restrict__ wswz, const int* __restrict__ edge_index,
    const float* __restrict__ alphaBuf, const float* __restrict__ denom,
    float* __restrict__ out)
{
    __shared__ v16bf wlds[W_CHUNKS];
    __shared__ float wgt[WAVES_PER_BLOCK][16][HEADS];
    __shared__ int   nodei[WAVES_PER_BLOCK][16];

    int wave = threadIdx.x >> 5, lane = threadIdx.x & 31;
    int tile = blockIdx.x * WAVES_PER_BLOCK + wave;
    bool active = tile < NTILES;
    int base = tile * 16;

    // wave0 kicks the W DMA; meanwhile all waves stage per-tile weights
    if (wave == 0) tdm_copy_64k((const void*)wswz, (void*)wlds);
    if (active && lane < 16) {
        int em = base + lane;
        int node = edge_index[em];
        nodei[wave][lane] = node;
        #pragma unroll
        for (int h = 0; h < HEADS; ++h)
            wgt[wave][lane][h] = alphaBuf[em * HEADS + h] /
                                 denom[node * HEADS + h] * 0.25f; // head mean
    }
    if (wave == 0) __builtin_amdgcn_s_wait_tensorcnt(0);
    __syncthreads();
    if (!active) return;

    int m = lane & 15;
    int e = base + m;
    int nj = edge_index[N_EDGES + e];
    v16bf aj[4];
    build_afrag(x, edge_attr, nj, e, lane, aj);

    int rowbase = (lane & 16) ? 8 : 0;
    #pragma unroll
    for (int t = 0; t < 4; ++t) {          // d-block: d = t*16 + lane&15
        float sum[8];
        #pragma unroll
        for (int r = 0; r < 8; ++r) sum[r] = 0.f;
        #pragma unroll
        for (int h = 0; h < HEADS; ++h) {  // sum heads in registers
            int nt = h * 4 + t;            // col = h*64 + t*16 + lane&15
            v8f cj = {};
            #pragma unroll
            for (int kk = 0; kk < 4; ++kk) {
                v16bf b = wlds[(nt * 4 + kk) * 32 + lane];
                cj = wmma_bf16(aj[kk], b, cj);
            }
            #pragma unroll
            for (int r = 0; r < 8; ++r)
                sum[r] += softplus_f(cj[r]) * wgt[wave][rowbase + r][h];
        }
        int d = t * 16 + (lane & 15);
        #pragma unroll
        for (int r = 0; r < 8; ++r)
            atomicAdd(&out[(size_t)nodei[wave][rowbase + r] * DIM + d], sum[r]);
    }
}

// ---- host-side launcher ----------------------------------------------------
extern "C" void kernel_launch(void* const* d_in, const int* in_sizes, int n_in,
                              void* d_out, int out_size, void* d_ws, size_t ws_size,
                              hipStream_t stream)
{
    const float* x         = (const float*)d_in[0];
    const float* edge_attr = (const float*)d_in[1];
    const float* W         = (const float*)d_in[2];
    const float* att       = (const float*)d_in[3];
    const float* bias      = (const float*)d_in[4];
    const float* bn_gamma  = (const float*)d_in[5];
    const float* bn_beta   = (const float*)d_in[6];
    const float* bn_mean   = (const float*)d_in[7];
    const float* bn_var    = (const float*)d_in[8];
    const int*   edge_index= (const int*)d_in[9];
    float* out = (float*)d_out;

    // workspace: alpha [E*H] f32 | amax bits [N*H] i32 | denom [N*H] f32
    //          | wswz 64 KB bf16 swizzled W  (~4.9 MB total)
    float*  alphaBuf = (float*)d_ws;
    int*    amaxBits = (int*)(alphaBuf + (size_t)N_EDGES * HEADS);
    float*  denom    = (float*)(amaxBits + (size_t)N_NODES * HEADS);
    bf16_t* wswz     = (bf16_t*)(denom + (size_t)N_NODES * HEADS);

    k_prep<<<(W_CHUNKS + BLOCK - 1) / BLOCK, BLOCK, 0, stream>>>(W, wswz);

    int initN = N_NODES * DIM;               // covers N*HEADS too
    k_init<<<(initN + BLOCK - 1) / BLOCK, BLOCK, 0, stream>>>(out, bias, amaxBits, denom);

    int blocks = (NTILES + WAVES_PER_BLOCK - 1) / WAVES_PER_BLOCK;
    k_alpha<<<blocks, BLOCK, 0, stream>>>(x, edge_attr, wswz, att,
                                          bn_gamma, bn_beta, bn_mean, bn_var,
                                          edge_index, alphaBuf, amaxBits);

    int nEH = N_EDGES * HEADS;
    k_norm<<<(nEH + BLOCK - 1) / BLOCK, BLOCK, 0, stream>>>(edge_index, alphaBuf,
                                                            amaxBits, denom);

    k_aggr<<<blocks, BLOCK, 0, stream>>>(x, edge_attr, wswz, edge_index,
                                         alphaBuf, denom, out);
}
